// KalmanNetNN_29197187678311
// MI455X (gfx1250) — matrix operand order repaired and checked by reading, not compiled, not run on hardware
//
#include <hip/hip_runtime.h>

typedef float  v2f   __attribute__((ext_vector_type(2)));
typedef float  v8f   __attribute__((ext_vector_type(8)));
typedef __bf16 v16bf __attribute__((ext_vector_type(16)));

__device__ __forceinline__ v8f wmma_f32x4(v2f a, v2f b, v8f c) {
  return __builtin_amdgcn_wmma_f32_16x16x4_f32(false, a, false, b, (short)0, c, false, false);
}
__device__ __forceinline__ v8f wmma_bf16x32(v16bf a, v16bf b, v8f c) {
  return __builtin_amdgcn_wmma_f32_16x16x32_bf16(false, a, false, b, (short)0, c, false, false);
}
__device__ __forceinline__ float sigmoidf_(float x) { return 1.0f / (1.0f + __expf(-x)); }
__device__ __forceinline__ v8f zero8() { v8f z = {0.f,0.f,0.f,0.f,0.f,0.f,0.f,0.f}; return z; }

// Accumulate C(16x16) += A(16 x kCount) * W[n0:n0+16, wk0:wk0+kCount]^T
// A row-major (LDS or global), rows indexed by m=lane&15, zero-padded up to kCount,
// 8-byte alignment guaranteed (even strides, even k offsets) -> vector b64 loads.
// rowGuard/kGuard are compile-time constants at each call site; guards are
// branchless (clamped index + select-zero) so loads stay unconditional.
__device__ __forceinline__ v8f gemm_any(v8f c, const float* A, int aStride,
                                        const float* W, int kdim, int outDim,
                                        int n0, int kCount, int wk0, int m, int half,
                                        bool rowGuard, bool kGuard) {
  const int row = n0 + m;
  const bool rowOK = !rowGuard || (row < outDim);
  const float* wrow = W + (size_t)(rowOK ? row : 0) * kdim + wk0;
  for (int k0 = 0; k0 < kCount; k0 += 4) {
    const int ka = k0 + 2 * half;
    const v2f a = *(const v2f*)(A + m * aStride + ka);
    v2f b;
    if (kGuard) {
      const int k1 = wk0 + ka, k2 = wk0 + ka + 1;
      const float bx = wrow[(k1 < kdim) ? ka : 0];
      const float by = wrow[(k2 < kdim) ? (ka + 1) : 0];
      b.x = (rowOK && k1 < kdim) ? bx : 0.0f;
      b.y = (rowOK && k2 < kdim) ? by : 0.0f;
    } else {
      b = *(const v2f*)(wrow + ka);
      if (rowGuard) {
        b.x = rowOK ? b.x : 0.0f;
        b.y = rowOK ? b.y : 0.0f;
      }
    }
    c = wmma_f32x4(a, b, c);
  }
  return c;
}

// Store C fragment (+bias, optional relu) to row-major buffer.
__device__ __forceinline__ void fc_store(v8f c, const float* bias, float* outp, int outStride,
                                         int outDim, int n0, int m, int half, bool doRelu) {
  const int col = n0 + m;
  if (col < outDim) {
    const float bv = bias[col];
#pragma unroll
    for (int r = 0; r < 8; ++r) {
      float v = c[r] + bv;
      if (doRelu) v = fmaxf(v, 0.0f);
      outp[(r + 8 * half) * outStride + col] = v;
    }
  }
}

// One GRU cell step for a 16-row tile, H=128, gates packed [r|z|n] in 3*128 rows.
// xKGuard: compile-time flag -> K-range guard on the input GEMMs only.
__device__ __forceinline__ void gru_run(const float* X, int xStride, int kCount, int kReal,
                                        const float* Hold, float* HnewLds, float* HnewGlob,
                                        const float* Wih, const float* Whh,
                                        const float* bih, const float* bhh,
                                        int m, int half, bool xKGuard) {
#pragma unroll 1
  for (int j = 0; j < 8; ++j) {
    const int n0 = j * 16;
    v8f ir = zero8(), iz = zero8(), in_ = zero8();
    v8f hr = zero8(), hz = zero8(), hn = zero8();
    ir  = gemm_any(ir,  X, xStride, Wih, kReal, 384, n0,       kCount, 0, m, half, false, xKGuard);
    iz  = gemm_any(iz,  X, xStride, Wih, kReal, 384, 128 + n0, kCount, 0, m, half, false, xKGuard);
    in_ = gemm_any(in_, X, xStride, Wih, kReal, 384, 256 + n0, kCount, 0, m, half, false, xKGuard);
    hr  = gemm_any(hr,  Hold, 128, Whh, 128, 384, n0,       128, 0, m, half, false, false);
    hz  = gemm_any(hz,  Hold, 128, Whh, 128, 384, 128 + n0, 128, 0, m, half, false, false);
    hn  = gemm_any(hn,  Hold, 128, Whh, 128, 384, 256 + n0, 128, 0, m, half, false, false);
    const int col = n0 + m;
    const float bir = bih[col],       bhr = bhh[col];
    const float biz = bih[128 + col], bhz = bhh[128 + col];
    const float bin = bih[256 + col], bhn = bhh[256 + col];
#pragma unroll
    for (int r = 0; r < 8; ++r) {
      const int row = r + 8 * half;
      const float hv = Hold[row * 128 + col];
      const float rg = sigmoidf_(ir[r] + bir + hr[r] + bhr);
      const float zg = sigmoidf_(iz[r] + biz + hz[r] + bhz);
      const float ng = tanhf(in_[r] + bin + rg * (hn[r] + bhn));
      const float hNew = (1.0f - zg) * ng + zg * hv;
      HnewLds[row * 128 + col] = hNew;
      if (HnewGlob) HnewGlob[row * 128 + col] = hNew;
    }
  }
}

struct KParams {
  const float *y, *u, *x_post, *h_Q, *h_Sigma, *h_S;
  const float *F, *G, *Hm, *innov_logit;
  const float *W_fc5, *b_fc5;
  const float *Wih_Q, *Whh_Q, *bih_Q, *bhh_Q;
  const float *Wih_Sig, *Whh_Sig, *bih_Sig, *bhh_Sig;
  const float *W_fc1, *b_fc1, *W_fc7, *b_fc7;
  const float *Wih_S, *Whh_S, *bih_S, *bhh_S;
  const float *W_fc2a, *b_fc2a, *W_fc2b, *b_fc2b;
  const float *W_fc3, *b_fc3, *W_fc4, *b_fc4;
  float *out_m1x, *out_hQ, *out_hSigNew, *out_hS;
};

__global__ __launch_bounds__(32) void kknet_kernel(KParams p) {
  __shared__ float s_xprior[16][4];
  __shared__ float s_dy[16][4];        // cols 2..3 zero pad
  __shared__ float s_out5[16][20];
  __shared__ float s_hQ[16][128];
  __shared__ float s_hSig[16][128];
  __shared__ float s_hS[16][128];
  __shared__ float s_sin[16][8];       // out1(4) | out7(2) | zero pad(2)
  __shared__ float s_out2[16][8];
  __shared__ float s_out3[16][16];
  __shared__ __align__(16) __bf16 s_in2[16][256];
  __shared__ float s_hbuf[16][16];

  const int lane = threadIdx.x & 31;
  const int m    = lane & 15;
  const int half = lane >> 4;
  const int b0   = blockIdx.x * 16;

  // ---------- step_prior (scalar, M=4/N=2) ----------
  for (int idx = lane; idx < 64; idx += 32) {
    const int row = idx >> 2, mm = idx & 3;
    float acc = 0.f;
    for (int k = 0; k < 4; ++k) acc += p.F[mm * 4 + k] * p.x_post[(size_t)(b0 + row) * 4 + k];
    acc += p.G[mm * 2 + 0] * p.u[(size_t)(b0 + row) * 2 + 0];
    acc += p.G[mm * 2 + 1] * p.u[(size_t)(b0 + row) * 2 + 1];
    s_xprior[row][mm] = acc;
  }
  if (lane < 16) {
    s_dy[lane][2] = 0.f; s_dy[lane][3] = 0.f;
    s_sin[lane][6] = 0.f; s_sin[lane][7] = 0.f;
  }
  __syncthreads();
  {
    const int row = lane >> 1, nn = lane & 1;
    float yp = 0.f;
    for (int k = 0; k < 4; ++k) yp += p.Hm[nn * 4 + k] * s_xprior[row][k];
    s_dy[row][nn] = p.y[(size_t)(b0 + row) * 2 + nn] - yp;
  }
  __syncthreads();

  // ---------- fc5: [16,4] -> relu [16,20] (tile0 unguarded, tile1 row-guarded) ----------
  {
    v8f c = zero8();
    c = gemm_any(c, &s_xprior[0][0], 4, p.W_fc5, 4, 20, 0, 4, 0, m, half, false, false);
    fc_store(c, p.b_fc5, &s_out5[0][0], 20, 20, 0, m, half, true);
  }
  {
    v8f c = zero8();
    c = gemm_any(c, &s_xprior[0][0], 4, p.W_fc5, 4, 20, 16, 4, 0, m, half, true, false);
    fc_store(c, p.b_fc5, &s_out5[0][0], 20, 20, 16, m, half, true);
  }
  __syncthreads();

  // ---------- GRU_Q (x=out5 K=20, h=h_Q) -> hQ_new (LDS + d_out) ----------
  gru_run(&s_out5[0][0], 20, 20, 20, p.h_Q + (size_t)b0 * 128, &s_hQ[0][0],
          p.out_hQ + (size_t)b0 * 128, p.Wih_Q, p.Whh_Q, p.bih_Q, p.bhh_Q, m, half, false);
  __syncthreads();

  // ---------- GRU_Sigma (x=hQ_new K=128, h=h_Sigma) -> hSig (LDS) ----------
  gru_run(&s_hQ[0][0], 128, 128, 128, p.h_Sigma + (size_t)b0 * 128, &s_hSig[0][0],
          nullptr, p.Wih_Sig, p.Whh_Sig, p.bih_Sig, p.bhh_Sig, m, half, false);
  __syncthreads();

  // ---------- fc1 (out1 -> s_sin[:,0:4]) and fc7 (out7 -> s_sin[:,4:6]) ----------
  {
    v8f c = zero8();
    c = gemm_any(c, &s_hSig[0][0], 128, p.W_fc1, 128, 4, 0, 128, 0, m, half, true, false);
    fc_store(c, p.b_fc1, &s_sin[0][0], 8, 4, 0, m, half, true);
  }
  {
    v8f c = zero8();
    c = gemm_any(c, &s_dy[0][0], 4, p.W_fc7, 2, 2, 0, 4, 0, m, half, true, true);
    fc_store(c, p.b_fc7, &s_sin[0][4], 8, 2, 0, m, half, true);
  }
  __syncthreads();

  // ---------- GRU_S (x=[out1|out7] K=6 pad8, h=h_S) -> hS_new (LDS + d_out) ----------
  gru_run(&s_sin[0][0], 8, 8, 6, p.h_S + (size_t)b0 * 128, &s_hS[0][0],
          p.out_hS + (size_t)b0 * 128, p.Wih_S, p.Whh_S, p.bih_S, p.bhh_S, m, half, true);
  __syncthreads();

  // ---------- fc2a (bf16 WMMA, K=256, 640 tiles) fused with fc2b accumulation ----------
  for (int i = lane; i < 16 * 256; i += 32) {
    const int r = i >> 8, cidx = i & 255;
    const float v = (cidx < 128) ? s_hSig[r][cidx] : s_hS[r][cidx - 128];
    s_in2[r][cidx] = (__bf16)v;
  }
  __syncthreads();

  // Preload all 8 A fragments (in2 row m, bf16 16x32 layout: lanes<16 hold K {0..15})
  v16bf afr[8];
#pragma unroll
  for (int kb = 0; kb < 8; ++kb) {
    const __bf16* rowp = &s_in2[m][0];
    const int k0 = kb * 32;
    union { v16bf v; uint4 q[2]; } u;
    u.q[0] = *(const uint4*)(rowp + k0 + half * 16);
    u.q[1] = *(const uint4*)(rowp + k0 + half * 16 + 8);
    afr[kb] = u.v;
  }

  v8f cOut = zero8();   // out2 accumulator [16 rows x 8 cols] (cols 8..15 garbage, masked on store)
#pragma unroll 1
  for (int t = 0; t < 640; ++t) {
    const int n0 = t * 16;
    v8f ch = zero8();
    const float* wbase = p.W_fc2a + (size_t)(n0 + m) * 256 + half * 16;
#pragma unroll
    for (int kb = 0; kb < 8; ++kb) {
      const float4* pw4 = (const float4*)(wbase + kb * 32);   // 64B-aligned
      v16bf bf;
#pragma unroll
      for (int q = 0; q < 4; ++q) {
        const float4 f = pw4[q];
        bf[q * 4 + 0] = (__bf16)f.x;
        bf[q * 4 + 1] = (__bf16)f.y;
        bf[q * 4 + 2] = (__bf16)f.z;
        bf[q * 4 + 3] = (__bf16)f.w;
      }
      ch = wmma_bf16x32(afr[kb], bf, ch);
    }
    // bias + relu, stage hidden chunk through LDS to transpose C-layout -> A-layout
    const float bv = p.b_fc2a[n0 + m];
#pragma unroll
    for (int r = 0; r < 8; ++r) s_hbuf[r + 8 * half][m] = fmaxf(ch[r] + bv, 0.0f);
    __syncthreads();
    cOut = gemm_any(cOut, &s_hbuf[0][0], 16, p.W_fc2b, 10240, 8, 0, 16, n0, m, half,
                    true, false);
    __syncthreads();
  }
  if (m < 8) {
    const float bv = p.b_fc2b[m];
#pragma unroll
    for (int r = 0; r < 8; ++r) s_out2[r + 8 * half][m] = cOut[r] + bv;
  }
  __syncthreads();

  // ---------- fc3: relu(W_fc3 @ [hS_new | out2]) -> [16,16] ----------
  {
    v8f c = zero8();
    c = gemm_any(c, &s_hS[0][0], 128, p.W_fc3, 136, 16, 0, 128, 0, m, half, false, false);
    c = gemm_any(c, &s_out2[0][0], 8, p.W_fc3, 136, 16, 0, 8, 128, m, half, false, false);
    fc_store(c, p.b_fc3, &s_out3[0][0], 16, 16, 0, m, half, true);
  }
  __syncthreads();

  // ---------- fc4: relu(W_fc4 @ [hSig | out3]) -> hSig_new (d_out) ----------
  {
    float* outSig = p.out_hSigNew + (size_t)b0 * 128;
#pragma unroll 1
    for (int t = 0; t < 8; ++t) {
      const int n0 = t * 16;
      v8f c = zero8();
      c = gemm_any(c, &s_hSig[0][0], 128, p.W_fc4, 144, 128, n0, 128, 0, m, half, false, false);
      c = gemm_any(c, &s_out3[0][0], 16, p.W_fc4, 144, 128, n0, 16, 128, m, half, false, false);
      const float bv = p.b_fc4[n0 + m];
#pragma unroll
      for (int r = 0; r < 8; ++r)
        outSig[(r + 8 * half) * 128 + n0 + m] = fmaxf(c[r] + bv, 0.0f);
    }
  }

  // ---------- posterior: m1x = x_prior + sigmoid(logit) * KG @ dy ----------
  const float gamma = sigmoidf_(p.innov_logit[0]);
  for (int idx = lane; idx < 64; idx += 32) {
    const int row = idx >> 2, mm = idx & 3;
    const float inov = s_out2[row][mm * 2 + 0] * s_dy[row][0] +
                       s_out2[row][mm * 2 + 1] * s_dy[row][1];
    p.out_m1x[(size_t)(b0 + row) * 4 + mm] = s_xprior[row][mm] + gamma * inov;
  }
}

extern "C" void kernel_launch(void* const* d_in, const int* in_sizes, int n_in,
                              void* d_out, int out_size, void* d_ws, size_t ws_size,
                              hipStream_t stream) {
  (void)n_in; (void)out_size; (void)d_ws; (void)ws_size;
  KParams p;
  p.y        = (const float*)d_in[0];
  p.u        = (const float*)d_in[1];
  p.x_post   = (const float*)d_in[2];
  p.h_Q      = (const float*)d_in[3];
  p.h_Sigma  = (const float*)d_in[4];
  p.h_S      = (const float*)d_in[5];
  p.F        = (const float*)d_in[6];
  p.G        = (const float*)d_in[7];
  p.Hm       = (const float*)d_in[8];
  p.innov_logit = (const float*)d_in[9];
  p.W_fc5    = (const float*)d_in[10]; p.b_fc5 = (const float*)d_in[11];
  p.Wih_Q    = (const float*)d_in[12]; p.Whh_Q = (const float*)d_in[13];
  p.bih_Q    = (const float*)d_in[14]; p.bhh_Q = (const float*)d_in[15];
  p.Wih_Sig  = (const float*)d_in[16]; p.Whh_Sig = (const float*)d_in[17];
  p.bih_Sig  = (const float*)d_in[18]; p.bhh_Sig = (const float*)d_in[19];
  p.W_fc1    = (const float*)d_in[20]; p.b_fc1 = (const float*)d_in[21];
  p.W_fc7    = (const float*)d_in[22]; p.b_fc7 = (const float*)d_in[23];
  p.Wih_S    = (const float*)d_in[24]; p.Whh_S = (const float*)d_in[25];
  p.bih_S    = (const float*)d_in[26]; p.bhh_S = (const float*)d_in[27];
  p.W_fc2a   = (const float*)d_in[28]; p.b_fc2a = (const float*)d_in[29];
  p.W_fc2b   = (const float*)d_in[30]; p.b_fc2b = (const float*)d_in[31];
  p.W_fc3    = (const float*)d_in[32]; p.b_fc3 = (const float*)d_in[33];
  p.W_fc4    = (const float*)d_in[34]; p.b_fc4 = (const float*)d_in[35];

  const int B = in_sizes[2] / 4;   // m1x_post is [B, M=4, 1]
  float* out = (float*)d_out;
  p.out_m1x     = out;                                    // [B,4]
  p.out_hQ      = out + (size_t)B * 4;                    // [B,128]
  p.out_hSigNew = p.out_hQ + (size_t)B * 128;             // [B,128]
  p.out_hS      = p.out_hSigNew + (size_t)B * 128;        // [B,128]

  const int tiles = B / 16;       // 512 waves, one 16-row tile each
  kknet_kernel<<<dim3(tiles), dim3(32), 0, stream>>>(p);
}
